// Decoder_22299470201391
// MI455X (gfx1250) — matrix operand order repaired and checked by reading, not compiled, not run on hardware
//
#include <hip/hip_runtime.h>
#include <hip/hip_bf16.h>

typedef __bf16 bf16_t;
typedef __attribute__((ext_vector_type(16))) __bf16 v16bf;
typedef __attribute__((ext_vector_type(8)))  float  v8f;

#define BQ      64
#define SEQ     256
#define DMODEL  512
#define NH      8
#define DHEAD   64
#define DFF     2048
#define NVOCAB  6000
#define NPACK   8128
#define NROWS   (BQ * SEQ)          // 16384

enum { ST_F32 = 0, ST_BF16 = 1, ST_RELU_BF16 = 2, ST_VT = 3 };

// ---------------- WMMA fragment loads (bf16, 16x16x32) ----------------
// A-matrix 16x32 bf16 (ISA 7.12.2): lane l<16 -> row M=l, K = {k0..k0+7, k0+16..k0+23}
//                                   lane l>=16 -> same row, K shifted by +8.
__device__ __forceinline__ v16bf load_fragA(const bf16_t* rowPtr, int k0, int hl) {
    const bf16_t* p = rowPtr + k0 + hl * 8;
    v16bf r;
    reinterpret_cast<float4*>(&r)[0] = *reinterpret_cast<const float4*>(p);
    reinterpret_cast<float4*>(&r)[1] = *reinterpret_cast<const float4*>(p + 16);
    return r;
}
// B-matrix 32x16 bf16: lane l holds column N=(l&15); K = {k0..k0+7, k0+16..k0+23},
// high lanes K shifted by +8 (same as A). rowPtr points at row n of W^T (ld = K).
__device__ __forceinline__ v16bf load_fragB(const bf16_t* rowPtr, int k0, int hl) {
    const bf16_t* p = rowPtr + k0 + hl * 8;
    v16bf r;
    reinterpret_cast<float4*>(&r)[0] = *reinterpret_cast<const float4*>(p);
    reinterpret_cast<float4*>(&r)[1] = *reinterpret_cast<const float4*>(p + 16);
    return r;
}

__device__ __forceinline__ v8f wmma_bf16(v16bf a, v16bf b, v8f c) {
    return __builtin_amdgcn_wmma_f32_16x16x32_bf16(false, a, false, b, (short)0, c, false, false);
}

// ---------------- input prep: bf16 convert + jnp.repeat(dec, 4, axis=1) ----------------
__global__ __launch_bounds__(256) void prep_inputs(const float* __restrict__ enc,
                                                   const float* __restrict__ dec,
                                                   bf16_t* __restrict__ enc16,
                                                   bf16_t* __restrict__ dec16) {
    long i = (long)blockIdx.x * blockDim.x + threadIdx.x;   // over NROWS*DMODEL
    if (i >= (long)NROWS * DMODEL) return;
    enc16[i] = (bf16_t)enc[i];
    long row = i >> 9, c = i & (DMODEL - 1);
    long b = row >> 8, s = row & (SEQ - 1);
    // jnp.repeat(x, 4, axis=1): out[b, s] = x[b, s // 4]
    dec16[i] = (bf16_t)dec[(b * 64 + (s >> 2)) * DMODEL + c];
}

// ---------------- weight transpose + bf16 convert: src[K][N] f32 -> dst[N][K] bf16 ---
__global__ __launch_bounds__(256) void transpose_w(const float* __restrict__ src,
                                                   bf16_t* __restrict__ dst, int K, int N) {
    long i = (long)blockIdx.x * blockDim.x + threadIdx.x;
    if (i >= (long)K * N) return;
    long n = i / K, k = i - n * K;
    dst[i] = (bf16_t)src[k * (long)N + n];
}

// ---------------- generic bf16 WMMA GEMM:  C[M,N] = A[M,K] @ W[K,N] ----------------
// Requires M%64==0, N%64==0, K%32==0. W passed as W^T bf16 [N][K].
// 4 waves/block, each wave computes a 16x64 strip (4 WMMA n-tiles), branchless k-loop.
__global__ __launch_bounds__(128) void gemm_bf16(const bf16_t* __restrict__ A, int lda,
                                                 const bf16_t* __restrict__ WT, int ldw,
                                                 float* __restrict__ outF,
                                                 bf16_t* __restrict__ outB,
                                                 int N, int K, int mode) {
    const int wave = threadIdx.x >> 5;
    const int lane = threadIdx.x & 31;
    const int l = lane & 15, hl = lane >> 4;
    const int m0 = (blockIdx.x * 4 + wave) * 16;
    const int n0 = blockIdx.y * 64;

    const bf16_t* aRow = A + (long)(m0 + l) * lda;
    const bf16_t* b0 = WT + (long)(n0 + l) * ldw;
    const bf16_t* b1 = WT + (long)(n0 + 16 + l) * ldw;
    const bf16_t* b2 = WT + (long)(n0 + 32 + l) * ldw;
    const bf16_t* b3 = WT + (long)(n0 + 48 + l) * ldw;

    v8f acc0 = {}, acc1 = {}, acc2 = {}, acc3 = {};
    for (int k0 = 0; k0 < K; k0 += 32) {
        __builtin_prefetch((const void*)(aRow + k0 + 256), 0, 1);   // global_prefetch_b8
        v16bf a = load_fragA(aRow, k0, hl);
        acc0 = wmma_bf16(a, load_fragB(b0, k0, hl), acc0);
        acc1 = wmma_bf16(a, load_fragB(b1, k0, hl), acc1);
        acc2 = wmma_bf16(a, load_fragB(b2, k0, hl), acc2);
        acc3 = wmma_bf16(a, load_fragB(b3, k0, hl), acc3);
    }

    // C layout: VGPR r -> lanes 0-15: (M=m0+r, N=n0..+15); lanes 16-31: (M=m0+8+r)
#pragma unroll
    for (int r = 0; r < 8; r++) {
        long m = m0 + r + hl * 8;
        float vals[4] = { acc0[r], acc1[r], acc2[r], acc3[r] };
#pragma unroll
        for (int s = 0; s < 4; s++) {
            long n = n0 + s * 16 + l;
            float val = vals[s];
            if (mode == ST_F32)            outF[m * (long)N + n] = val;
            else if (mode == ST_BF16)      outB[m * (long)N + n] = (bf16_t)val;
            else if (mode == ST_RELU_BF16) outB[m * (long)N + n] = (bf16_t)fmaxf(val, 0.0f);
            else { // ST_VT: m=(b,s), n=(h,dh) -> vT[b][h][dh][s]
                long bb = m >> 8, ss = m & (SEQ - 1);
                long hh = n >> 6, dd = n & (DHEAD - 1);
                outB[((bb * NH + hh) * DHEAD + dd) * SEQ + ss] = (bf16_t)val;
            }
        }
    }
}

// ---------------- fused attention: one wave per (b, h, 16-row q tile) ----------------
// Q,K: bf16 [NROWS, DMODEL] (head slice at column h*64); VT: bf16 [B,H,DHEAD,SEQ]
// O: bf16 [NROWS, DMODEL]
__global__ __launch_bounds__(32) void attention_kernel(const bf16_t* __restrict__ Q,
                                                       const bf16_t* __restrict__ Km,
                                                       const bf16_t* __restrict__ VT,
                                                       bf16_t* __restrict__ O, int causal) {
    __shared__ __align__(16) float  sc[16][SEQ];
    __shared__ __align__(16) bf16_t pb[16][SEQ];
    __shared__ float invs[16];

    const int lane = threadIdx.x;
    const int l = lane & 15, hl = lane >> 4;
    const int qt = blockIdx.x, h = blockIdx.y, b = blockIdx.z;
    const long bs = (long)b * SEQ;

    // ---- phase 1: scores = (Q K^T) / 8 with causal mask -> LDS
    const bf16_t* qRow = Q + (bs + qt * 16 + l) * DMODEL + h * DHEAD;
    for (int kt = 0; kt < SEQ / 16; kt++) {
        const bf16_t* kRow = Km + (bs + kt * 16 + l) * DMODEL + h * DHEAD;
        v8f acc = {};
        acc = wmma_bf16(load_fragA(qRow, 0, hl),  load_fragB(kRow, 0, hl),  acc);
        acc = wmma_bf16(load_fragA(qRow, 32, hl), load_fragB(kRow, 32, hl), acc);
        int col = kt * 16 + l;
#pragma unroll
        for (int r = 0; r < 8; r++) {
            int m = r + hl * 8;
            float v = acc[r] * 0.125f;                 // 1/sqrt(64)
            if (causal && col > qt * 16 + m) v = -1e9f;
            sc[m][col] = v;
        }
    }
    __syncthreads();

    // ---- phase 2: row softmax (store unnormalized exp as bf16, keep 1/sum)
    for (int m = 0; m < 16; m++) {
        float mx = -1e30f;
        for (int j = lane; j < SEQ; j += 32) mx = fmaxf(mx, sc[m][j]);
        for (int off = 16; off; off >>= 1) mx = fmaxf(mx, __shfl_xor(mx, off, 32));
        float sm = 0.0f;
        for (int j = lane; j < SEQ; j += 32) {
            float e = __expf(sc[m][j] - mx);
            sm += e;
            pb[m][j] = (bf16_t)e;
        }
        for (int off = 16; off; off >>= 1) sm += __shfl_xor(sm, off, 32);
        if (lane == 0) invs[m] = 1.0f / sm;
    }
    __syncthreads();

    // ---- phase 3: O = P @ V  (B-frags from V^T), scale by 1/sum at store
    const long bh = (long)b * NH + h;
    for (int nt = 0; nt < DHEAD / 16; nt++) {
        const bf16_t* aRow = &pb[l][0];
        const bf16_t* vRow = VT + (bh * DHEAD + nt * 16 + l) * SEQ;
        v8f acc = {};
        for (int k0 = 0; k0 < SEQ; k0 += 32)
            acc = wmma_bf16(load_fragA(aRow, k0, hl), load_fragB(vRow, k0, hl), acc);
#pragma unroll
        for (int r = 0; r < 8; r++) {
            int m = r + hl * 8;
            O[(bs + qt * 16 + m) * DMODEL + h * DHEAD + nt * 16 + l] =
                (bf16_t)(acc[r] * invs[m]);
        }
    }
}

// ---------------- residual add + LayerNorm (one wave per row) ----------------
__global__ __launch_bounds__(256) void add_ln(const float* __restrict__ resid,
                                              const float* __restrict__ dec_in,
                                              const float* __restrict__ addv,
                                              const float* __restrict__ g,
                                              const float* __restrict__ beta,
                                              float* __restrict__ out32,
                                              bf16_t* __restrict__ out16, int dec_mode) {
    int row = (int)((blockIdx.x * blockDim.x + threadIdx.x) >> 5);
    int lane = threadIdx.x & 31;
    if (row >= NROWS) return;
    const float* rp;
    if (dec_mode) {
        int b = row >> 8, s = row & (SEQ - 1);
        rp = dec_in + ((long)b * 64 + (s >> 2)) * DMODEL;   // repeated decoder input
    } else {
        rp = resid + (long)row * DMODEL;
    }
    const float* ap = addv + (long)row * DMODEL;
    float x[16], sum = 0.0f, sq = 0.0f;
#pragma unroll
    for (int i = 0; i < 16; i++) {
        float v = rp[lane + 32 * i] + ap[lane + 32 * i];
        x[i] = v; sum += v; sq += v * v;
    }
    for (int off = 16; off; off >>= 1) {
        sum += __shfl_xor(sum, off, 32);
        sq  += __shfl_xor(sq,  off, 32);
    }
    float mean = sum * (1.0f / DMODEL);
    float var  = sq * (1.0f / DMODEL) - mean * mean;
    float rs   = rsqrtf(var + 1e-6f);
#pragma unroll
    for (int i = 0; i < 16; i++) {
        int c = lane + 32 * i;
        float v = (x[i] - mean) * rs * g[c] + beta[c];
        out32[(long)row * DMODEL + c] = v;
        out16[(long)row * DMODEL + c] = (bf16_t)v;
    }
}

// ------- fused: ragged gather + vocab GEMM + bias + online softmax --------
// One block = 16 packed rows. 8 waves split the 375 n-tiles; each keeps
// per-row running (max, sum); reduced lane-group -> wave -> block; then the
// prob pass re-reads this block's own 384 KB of logits (L2-hot) and streams
// prob with non-temporal stores.
__global__ __launch_bounds__(256) void logits_softmax(const bf16_t* __restrict__ X,
                                                      const int* __restrict__ gather,
                                                      const bf16_t* __restrict__ WT,
                                                      const float* __restrict__ bias,
                                                      float* __restrict__ logits,
                                                      float* __restrict__ prob) {
    __shared__ float red_m[8][16], red_s[8][16];
    __shared__ float rowM[16], rowInv[16];

    const int wave = threadIdx.x >> 5;
    const int lane = threadIdx.x & 31;
    const int l = lane & 15, hl = lane >> 4;
    const int m0 = blockIdx.x * 16;

    const bf16_t* aRow = X + (long)gather[m0 + l] * DMODEL;

    float mloc[8], sloc[8];
#pragma unroll
    for (int r = 0; r < 8; r++) { mloc[r] = -1e30f; sloc[r] = 0.0f; }

    for (int nt = wave; nt < NVOCAB / 16; nt += 8) {
        const bf16_t* bRow = WT + (long)(nt * 16 + l) * DMODEL;
        v8f acc = {};
        for (int k0 = 0; k0 < DMODEL; k0 += 32)
            acc = wmma_bf16(load_fragA(aRow, k0, hl), load_fragB(bRow, k0, hl), acc);
        long n = nt * 16 + l;
        float bval = bias[n];
#pragma unroll
        for (int r = 0; r < 8; r++) {
            long m = m0 + r + hl * 8;
            float v = acc[r] + bval;
            logits[m * NVOCAB + n] = v;
            float nm = fmaxf(mloc[r], v);
            sloc[r] = sloc[r] * __expf(mloc[r] - nm) + __expf(v - nm);
            mloc[r] = nm;
        }
    }
    // reduce within the 16-lane group that shares rows (r + hl*8)
#pragma unroll
    for (int r = 0; r < 8; r++) {
        float m = mloc[r], s = sloc[r];
        for (int off = 1; off < 16; off <<= 1) {
            float mo = __shfl_xor(m, off, 16);
            float so = __shfl_xor(s, off, 16);
            float nm = fmaxf(m, mo);
            s = s * __expf(m - nm) + so * __expf(mo - nm);
            m = nm;
        }
        if (l == 0) { red_m[wave][r + hl * 8] = m; red_s[wave][r + hl * 8] = s; }
    }
    __threadfence_block();
    __syncthreads();
    if (threadIdx.x < 16) {
        float m = -1e30f, s = 0.0f;
#pragma unroll
        for (int w = 0; w < 8; w++) {
            float mo = red_m[w][threadIdx.x], so = red_s[w][threadIdx.x];
            float nm = fmaxf(m, mo);
            s = s * __expf(m - nm) + so * __expf(mo - nm);
            m = nm;
        }
        rowM[threadIdx.x] = m;
        rowInv[threadIdx.x] = 1.0f / s;
    }
    __syncthreads();
    // prob pass: logits rows just written by this block -> L2-hot re-read
    for (int r = 0; r < 16; r++) {
        float M = rowM[r], I = rowInv[r];
        const float* lp = logits + (long)(m0 + r) * NVOCAB;
        float* pp = prob + (long)(m0 + r) * NVOCAB;
        for (int j = threadIdx.x; j < NVOCAB; j += 256)
            __builtin_nontemporal_store(__expf(lp[j] - M) * I, &pp[j]);
    }
}

// =====================================================================
extern "C" void kernel_launch(void* const* d_in, const int* in_sizes, int n_in,
                              void* d_out, int out_size, void* d_ws, size_t ws_size,
                              hipStream_t stream) {
    const float* enc    = (const float*)d_in[0];
    const float* dec    = (const float*)d_in[1];
    const float* w_s[4] = { (const float*)d_in[2], (const float*)d_in[3],
                            (const float*)d_in[4], (const float*)d_in[5] };
    const float* w_x[4] = { (const float*)d_in[6], (const float*)d_in[7],
                            (const float*)d_in[8], (const float*)d_in[9] };
    const float* ffn_w1 = (const float*)d_in[10];
    const float* ffn_w2 = (const float*)d_in[11];
    const float* ln_g[3] = { (const float*)d_in[12], (const float*)d_in[14], (const float*)d_in[16] };
    const float* ln_b[3] = { (const float*)d_in[13], (const float*)d_in[15], (const float*)d_in[17] };
    const float* out_w  = (const float*)d_in[18];
    const float* out_b  = (const float*)d_in[19];
    const int*   pack   = (const int*)d_in[21];
    (void)in_sizes; (void)n_in; (void)out_size; (void)ws_size;

    // ---- workspace layout
    char* base = (char*)d_ws;
    size_t off = 0;
    auto alloc = [&](size_t bytes) -> char* {
        char* p = base + off;
        off = (off + bytes + 255) & ~(size_t)255;
        return p;
    };
    const size_t W512 = (size_t)DMODEL * DMODEL * sizeof(bf16_t);
    bf16_t* wT_s[4]; for (int i = 0; i < 4; i++) wT_s[i] = (bf16_t*)alloc(W512);
    bf16_t* wT_x[4]; for (int i = 0; i < 4; i++) wT_x[i] = (bf16_t*)alloc(W512);
    bf16_t* w1T  = (bf16_t*)alloc((size_t)DFF * DMODEL * sizeof(bf16_t));
    bf16_t* w2T  = (bf16_t*)alloc((size_t)DMODEL * DFF * sizeof(bf16_t));
    bf16_t* outT = (bf16_t*)alloc((size_t)NVOCAB * DMODEL * sizeof(bf16_t));

    const size_t ACT16 = (size_t)NROWS * DMODEL * sizeof(bf16_t);   // 16 MB
    const size_t ACT32 = (size_t)NROWS * DMODEL * sizeof(float);    // 32 MB
    bf16_t* dec16 = (bf16_t*)alloc(ACT16);
    bf16_t* enc16 = (bf16_t*)alloc(ACT16);
    char*   hreg  = alloc((size_t)NROWS * DFF * sizeof(bf16_t));    // 64 MB, aliased
    bf16_t* q16   = (bf16_t*)(hreg + 0 * ACT16);
    bf16_t* k16   = (bf16_t*)(hreg + 1 * ACT16);
    bf16_t* vT16  = (bf16_t*)(hreg + 2 * ACT16);
    bf16_t* o16   = (bf16_t*)(hreg + 3 * ACT16);
    bf16_t* h16   = (bf16_t*)hreg;                                  // FFN hidden (after attn)
    float*  c32   = (float*)alloc(ACT32);
    float*  x32a  = (float*)alloc(ACT32);
    bf16_t* x16a  = (bf16_t*)alloc(ACT16);
    float*  x32b  = (float*)alloc(ACT32);
    bf16_t* x16b  = (bf16_t*)alloc(ACT16);

    float* out_prob   = (float*)d_out;
    float* out_logits = out_prob + (size_t)NPACK * NVOCAB;
    float* out_decout = out_logits + (size_t)NPACK * NVOCAB;

    // ---- stage A: prep + weight transposes
    {
        long tot = (long)NROWS * DMODEL;
        prep_inputs<<<dim3((tot + 255) / 256), dim3(256), 0, stream>>>(enc, dec, enc16, dec16);
    }
    auto trans = [&](const float* src, bf16_t* dst, int K, int N) {
        long tot = (long)K * N;
        transpose_w<<<dim3((tot + 255) / 256), dim3(256), 0, stream>>>(src, dst, K, N);
    };
    for (int i = 0; i < 4; i++) trans(w_s[i], wT_s[i], DMODEL, DMODEL);
    for (int i = 0; i < 4; i++) trans(w_x[i], wT_x[i], DMODEL, DMODEL);
    trans(ffn_w1, w1T, DMODEL, DFF);
    trans(ffn_w2, w2T, DFF, DMODEL);
    trans(out_w, outT, DMODEL, NVOCAB);

    auto gemm = [&](const bf16_t* A, int lda, const bf16_t* WT, int ldw,
                    float* oF, bf16_t* oB, int M, int N, int K, int mode) {
        dim3 g(M / 64, N / 64);
        gemm_bf16<<<g, dim3(128), 0, stream>>>(A, lda, WT, ldw, oF, oB, N, K, mode);
    };
    dim3 attn_grid(SEQ / 16, NH, BQ);

    // ---- stage B: causal self-attention
    gemm(dec16, DMODEL, wT_s[0], DMODEL, nullptr, q16, NROWS, DMODEL, DMODEL, ST_BF16);
    gemm(dec16, DMODEL, wT_s[1], DMODEL, nullptr, k16, NROWS, DMODEL, DMODEL, ST_BF16);
    gemm(dec16, DMODEL, wT_s[2], DMODEL, nullptr, vT16, NROWS, DMODEL, DMODEL, ST_VT);
    attention_kernel<<<attn_grid, dim3(32), 0, stream>>>(q16, k16, vT16, o16, 1);
    gemm(o16, DMODEL, wT_s[3], DMODEL, c32, nullptr, NROWS, DMODEL, DMODEL, ST_F32);
    add_ln<<<dim3(NROWS / 8), dim3(256), 0, stream>>>(nullptr, dec, c32, ln_g[0], ln_b[0], x32a, x16a, 1);

    // ---- stage C: cross-attention (q from x1, kv from encoder)
    gemm(x16a, DMODEL, wT_x[0], DMODEL, nullptr, q16, NROWS, DMODEL, DMODEL, ST_BF16);
    gemm(enc16, DMODEL, wT_x[1], DMODEL, nullptr, k16, NROWS, DMODEL, DMODEL, ST_BF16);
    gemm(enc16, DMODEL, wT_x[2], DMODEL, nullptr, vT16, NROWS, DMODEL, DMODEL, ST_VT);
    attention_kernel<<<attn_grid, dim3(32), 0, stream>>>(q16, k16, vT16, o16, 0);
    gemm(o16, DMODEL, wT_x[3], DMODEL, c32, nullptr, NROWS, DMODEL, DMODEL, ST_F32);
    add_ln<<<dim3(NROWS / 8), dim3(256), 0, stream>>>(x32a, nullptr, c32, ln_g[1], ln_b[1], x32b, x16b, 0);

    // ---- stage D: FFN (h16 aliases q16..o16, which are dead now)
    gemm(x16b, DMODEL, w1T, DMODEL, nullptr, h16, NROWS, DFF, DMODEL, ST_RELU_BF16);
    gemm(h16, DFF, w2T, DFF, c32, nullptr, NROWS, DMODEL, DFF, ST_F32);
    add_ln<<<dim3(NROWS / 8), dim3(256), 0, stream>>>(x32b, nullptr, c32, ln_g[2], ln_b[2], out_decout, x16a, 0);

    // ---- stage E: fused ragged gather + vocab GEMM + bias + softmax
    logits_softmax<<<dim3(NPACK / 16), dim3(256), 0, stream>>>(x16a, pack, outT, out_b,
                                                               out_logits, out_prob);
}
